// Net_27865747816547
// MI455X (gfx1250) — matrix operand order repaired and checked by reading, not compiled, not run on hardware
//
#include <hip/hip_runtime.h>
#include <hip/hip_bf16.h>
#include <math.h>

#ifndef __has_builtin
#define __has_builtin(x) 0
#endif

typedef __attribute__((ext_vector_type(16))) _Float16 v16h;
typedef __attribute__((ext_vector_type(8)))  _Float16 v8h;
typedef __attribute__((ext_vector_type(8)))  float    v8f;
typedef __attribute__((ext_vector_type(2)))  float    v2f;

#define N0V   4096
#define FINV  1433
#define HIDV  32
#define NCLSV 16
#define NEV   131072
#define KL1   2000
#define KL2   1000
#define KL3   500

static __device__ __forceinline__ float wave_reduce_sum(float v) {
  for (int o = 16; o > 0; o >>= 1) v += __shfl_xor(v, o, 32);
  return v;
}

// ---------------------------------------------------------------- elementwise
__global__ void k_fill_zero(float* __restrict__ p, int count) {
  int i = blockIdx.x * blockDim.x + threadIdx.x;
  if (i < count) p[i] = 0.0f;
}

__global__ void k_copy(const float* __restrict__ src, float* __restrict__ dst, int count) {
  int i = blockIdx.x * blockDim.x + threadIdx.x;
  if (i < count) dst[i] = src[i];
}

// edge_index is int64 (2 x E, row-major). Drop self loops (reference zeroes diag).
__global__ void k_scatter_edges(float* __restrict__ A, const long long* __restrict__ ei,
                                int ne, int n) {
  int e = blockIdx.x * blockDim.x + threadIdx.x;
  if (e >= ne) return;
  int r = (int)ei[e];
  int c = (int)ei[(size_t)ne + e];
  if (r != c) atomicAdd(A + (size_t)r * n + c, 1.0f);
}

// ---------------------------------------------------------------- GCN pieces
// XW = X @ W   (wave per row, lane = output feature, shfl-broadcast of X)
__global__ void k_mm_xw(const float* __restrict__ X, const float* __restrict__ W,
                        float* __restrict__ out, int n, int fin, int fout) {
  int gid  = blockIdx.x * blockDim.x + threadIdx.x;
  int row  = gid >> 5;
  if (row >= n) return;
  int lane = threadIdx.x & 31;
  const float* xrow = X + (size_t)row * fin;
  float acc = 0.0f;
  for (int k0 = 0; k0 < fin; k0 += 32) {
    float xv = (k0 + lane < fin) ? xrow[k0 + lane] : 0.0f;
    int lim = fin - k0; if (lim > 32) lim = 32;
    for (int t = 0; t < lim; ++t) {
      float a = __shfl(xv, t, 32);
      if (lane < fout) acc += a * W[(size_t)(k0 + t) * fout + lane];
    }
  }
  if (lane < fout) out[(size_t)row * fout + lane] = acc;
}

// d[i] = rsqrt(rowsum(A) + 2)   (improved GCN: self-loop weight 2; deg always > 0)
__global__ void k_deg_rsqrt(const float* __restrict__ A, int n, float* __restrict__ d) {
  int gid = blockIdx.x * blockDim.x + threadIdx.x;
  int row = gid >> 5;
  if (row >= n) return;
  int lane = threadIdx.x & 31;
  const float* ar = A + (size_t)row * n;
  float s = 0.0f;
  for (int j = lane; j < n; j += 32) s += ar[j];
  s = wave_reduce_sum(s);
  if (lane == 0) d[row] = rsqrtf(s + 2.0f);
}

__global__ void k_scale_rows(const float* __restrict__ XW, const float* __restrict__ d,
                             float* __restrict__ XWs, int n, int fout) {
  int i = blockIdx.x * blockDim.x + threadIdx.x;
  if (i >= n * fout) return;
  XWs[i] = XW[i] * d[i / fout];
}

// Y = A @ XWs via V_WMMA_F32_16X16X4_F32. One wave per 16-row strip; the A fragment
// is reused across both N-tiles (TWO => fout==32) with independent accumulators so
// the matrix pipe keeps two WMMAs in flight. Templated on TWO so the inner loop is
// branch-free. VALU fallback if the builtin is missing.
template <bool TWO>
__global__ void k_spmm_tile16(const float* __restrict__ A, const float* __restrict__ B,
                              float* __restrict__ Y, int n) {
  const int fout = TWO ? 32 : 16;
  int wave = (blockIdx.x * blockDim.x + threadIdx.x) >> 5;
  if (wave >= (n >> 4)) return;            // wave-uniform: EXEC stays all-ones
  int tm = wave;
  int lane = threadIdx.x & 31;
  int hi = lane >> 4, r = lane & 15;
#if __has_builtin(__builtin_amdgcn_wmma_f32_16x16x4_f32)
  // A frag (16x4 f32): lanes0-15 hold M=lane, VGPR0=K0 VGPR1=K1; lanes16-31 K2/K3.
  const float* arow = A + (size_t)(tm * 16 + r) * n;
  v8f acc0 = {}, acc1 = {};
  for (int k0 = 0; k0 < n; k0 += 4) {
    int ka = k0 + hi * 2;
    v2f a, b;
    a[0] = arow[ka];
    a[1] = arow[ka + 1];
    b[0] = B[(size_t)ka * fout + r];
    b[1] = B[(size_t)(ka + 1) * fout + r];
    acc0 = __builtin_amdgcn_wmma_f32_16x16x4_f32(false, a, false, b, (short)0, acc0,
                                                 false, false);
    if (TWO) {
      v2f c;
      c[0] = B[(size_t)ka * fout + 16 + r];
      c[1] = B[(size_t)(ka + 1) * fout + 16 + r];
      acc1 = __builtin_amdgcn_wmma_f32_16x16x4_f32(false, a, false, c, (short)0, acc1,
                                                   false, false);
    }
  }
#pragma unroll
  for (int t = 0; t < 8; ++t) {            // C/D: VGPR t -> M = t + 8*hi, N = r
    Y[(size_t)(tm * 16 + t + hi * 8) * fout + r] = acc0[t];
    if (TWO) Y[(size_t)(tm * 16 + t + hi * 8) * fout + 16 + r] = acc1[t];
  }
#else
  const int tilesN = TWO ? 2 : 1;
  for (int tn = 0; tn < tilesN; ++tn) {
    int col = tn * 16 + r;
    for (int t = 0; t < 8; ++t) {
      int row = tm * 16 + t + hi * 8;
      const float* ar = A + (size_t)row * n;
      float acc = 0.0f;
      for (int j = 0; j < n; ++j) acc += ar[j] * B[(size_t)j * fout + col];
      Y[(size_t)row * fout + col] = acc;
    }
  }
#endif
}

// VALU fallback spmm for n not a multiple of 16 (n = 1000, 500)
__global__ void k_spmm_valu(const float* __restrict__ A, const float* __restrict__ B,
                            float* __restrict__ Y, int n, int fout) {
  int gid = blockIdx.x * blockDim.x + threadIdx.x;
  int row = gid >> 5;
  if (row >= n) return;
  int lane = threadIdx.x & 31;
  const float* ar = A + (size_t)row * n;
  float acc = 0.0f;
  for (int j0 = 0; j0 < n; j0 += 32) {
    float av = (j0 + lane < n) ? ar[j0 + lane] : 0.0f;
    int lim = n - j0; if (lim > 32) lim = 32;
    for (int t = 0; t < lim; ++t) {
      float a = __shfl(av, t, 32);
      if (lane < fout) acc += a * B[(size_t)(j0 + t) * fout + lane];
    }
  }
  if (lane < fout) Y[(size_t)row * fout + lane] = acc;
}

// out = d*Y + 2*d^2*XW + b   (= D (A+2I) D XW + b), optional relu
__global__ void k_combine_gcn(const float* __restrict__ Y, const float* __restrict__ XW,
                              const float* __restrict__ d, const float* __restrict__ b,
                              float* __restrict__ out, int n, int fout, int relu) {
  int i = blockIdx.x * blockDim.x + threadIdx.x;
  if (i >= n * fout) return;
  int row = i / fout, f = i - row * fout;
  float dv = d[row];
  float v = dv * Y[i] + 2.0f * dv * dv * XW[i] + b[f];
  out[i] = relu ? fmaxf(v, 0.0f) : v;
}

// ---------------------------------------------------------------- top-k pool
__global__ void k_score(const float* __restrict__ x, const float* __restrict__ p,
                        float* __restrict__ score, int n) {
  int gid = blockIdx.x * blockDim.x + threadIdx.x;
  int row = gid >> 5;
  if (row >= n) return;
  int lane = threadIdx.x & 31;
  float v = x[(size_t)row * HIDV + lane] * p[lane];
  v = wave_reduce_sum(v);
  if (lane == 0) {
    float ns = 0.0f;
    for (int i = 0; i < HIDV; ++i) { float pv = p[i]; ns += pv * pv; }
    score[row] = tanhf(v * rsqrtf(ns));
  }
}

// Single-workgroup bitonic sort (descending; ties -> lower index first, matching
// jax.lax.top_k). P is the padded pow2 size (<= 4096). 32KB LDS of 320KB WGP LDS.
__global__ void k_topk_sort(const float* __restrict__ score, int n, int P, int k,
                            int* __restrict__ perm) {
  __shared__ float key[4096];
  __shared__ int   idx[4096];
  int tid = threadIdx.x, bsz = blockDim.x;
  for (int i = tid; i < P; i += bsz) {
    key[i] = (i < n) ? score[i] : -3.4e38f;
    idx[i] = i;
  }
  __syncthreads();
  for (int kk = 2; kk <= P; kk <<= 1) {
    for (int j = kk >> 1; j > 0; j >>= 1) {
      for (int i = tid; i < P; i += bsz) {
        int l = i ^ j;
        if (l > i) {
          float ki = key[i], kl = key[l];
          int ii = idx[i], il = idx[l];
          bool iBefore = (ki > kl) || (ki == kl && ii < il);
          bool up = ((i & kk) == 0);      // up-blocks sort descending
          if (up ? !iBefore : iBefore) {
            key[i] = kl; key[l] = ki; idx[i] = il; idx[l] = ii;
          }
        }
      }
      __syncthreads();
    }
  }
  for (int r = tid; r < k; r += bsz) perm[r] = idx[r];
}

__global__ void k_gather_x(const float* __restrict__ x, const int* __restrict__ perm,
                           const float* __restrict__ score, int k, float* __restrict__ xp) {
  int i = blockIdx.x * blockDim.x + threadIdx.x;
  if (i >= k * HIDV) return;
  int r = i >> 5, f = i & 31;
  int pi = perm[r];
  xp[i] = x[(size_t)pi * HIDV + f] * score[pi];
}

// ---------------------------------------------------------------- fused augment+pool
// Ah  (Mp2 x Kp, f16) : row i = (A+I)[perm[i], :]   (zero padded; counts exact in f16)
// AhT (Np4 x Kp, f16) : row j = (A+I)[:, perm[j]]^T (zero padded)
__global__ void k_gather_rows_f16(const float* __restrict__ A, const int* __restrict__ perm,
                                  int n, int k, int Mp, int Kp, _Float16* __restrict__ Ah) {
  int i = blockIdx.x * blockDim.x + threadIdx.x;
  if (i >= Mp * Kp) return;
  int r = i / Kp, kk = i - r * Kp;
  float v = 0.0f;
  if (r < k && kk < n) {
    int pi = perm[r];
    v = A[(size_t)pi * n + kk] + (kk == pi ? 1.0f : 0.0f);
  }
  Ah[i] = (_Float16)v;
}

__global__ void k_gather_cols_f16(const float* __restrict__ A, const int* __restrict__ perm,
                                  int n, int k, int Mp, int Kp, _Float16* __restrict__ AhT) {
  int i = blockIdx.x * blockDim.x + threadIdx.x;
  if (i >= Mp * Kp) return;
  int c = i / Kp, kk = i - c * Kp;
  float v = 0.0f;
  if (c < k && kk < n) {
    int pj = perm[c];
    v = A[(size_t)kk * n + pj] + (kk == pj ? 1.0f : 0.0f);
  }
  AhT[i] = (_Float16)v;
}

// Apool = ((A+I)@(A+I))[perm][:,perm] with zeroed diagonal.
// Register-blocked 32x64 output per wave: 2 A fragments x 4 B fragments feed 8
// independent V_WMMA_F32_16X16X32_F16 per K-step (A reused 4x, B 2x; no D->A/B
// hazards between consecutive WMMAs). All fragment loads are contiguous b128; the
// inner loop is branch-free (prefetch is speculative, OOB prefetch is dropped and
// stays inside the workspace). ISA layouts: A 16-bit frag lanes0-15 K{0..7,16..23},
// lanes16-31 K{8..15,24..31}; B frag uses the same K order as A.
__global__ void k_augpool_gemm_f16(const _Float16* __restrict__ Ah,
                                   const _Float16* __restrict__ BhT,
                                   float* __restrict__ C, int k, int Kp,
                                   int mStrips, int nStrips) {
  int wave = (blockIdx.x * blockDim.x + threadIdx.x) >> 5;
  if (wave >= mStrips * nStrips) return;   // wave-uniform exit, EXEC all-ones for WMMA
  int sm = wave / nStrips, sn = wave % nStrips;
  int lane = threadIdx.x & 31;
  int hi = lane >> 4, r = lane & 15;
  const _Float16* arow = Ah  + (size_t)(sm * 32 + r) * Kp;
  const _Float16* brow = BhT + (size_t)(sn * 64 + r) * Kp;
  int aoff0 = hi * 8, aoff1 = 16 + hi * 8, boff = hi * 8;
  union HU { v16h v; v8h h[2]; };
  v8f acc[2][4];
#pragma unroll
  for (int mt = 0; mt < 2; ++mt)
#pragma unroll
    for (int nt = 0; nt < 4; ++nt) acc[mt][nt] = (v8f){};
  for (int k0 = 0; k0 < Kp; k0 += 32) {
    HU a[2], b[4];
#pragma unroll
    for (int mt = 0; mt < 2; ++mt) {
      const _Float16* ar = arow + (size_t)(mt * 16) * Kp + k0;
      a[mt].h[0] = *(const v8h*)(ar + aoff0);   // 16B aligned: Kp%32==0
      a[mt].h[1] = *(const v8h*)(ar + aoff1);
      __builtin_prefetch(ar + 32, 0, 3);        // gfx1250 global_prefetch_b8
    }
#pragma unroll
    for (int nt = 0; nt < 4; ++nt) {
      const _Float16* bc = brow + (size_t)(nt * 16) * Kp + k0;
      b[nt].h[0] = *(const v8h*)(bc + boff);
      b[nt].h[1] = *(const v8h*)(bc + boff + 16);
      __builtin_prefetch(bc + 32, 0, 3);
    }
#pragma unroll
    for (int nt = 0; nt < 4; ++nt)
#pragma unroll
      for (int mt = 0; mt < 2; ++mt)
        acc[mt][nt] = __builtin_amdgcn_wmma_f32_16x16x32_f16(
            false, a[mt].v, false, b[nt].v, (short)0, acc[mt][nt], false, false);
  }
#pragma unroll
  for (int mt = 0; mt < 2; ++mt)
#pragma unroll
    for (int nt = 0; nt < 4; ++nt)
#pragma unroll
      for (int t = 0; t < 8; ++t) {
        int row = sm * 32 + mt * 16 + t + hi * 8;
        int col = sn * 64 + nt * 16 + r;
        if (row < k && col < k)
          C[(size_t)row * k + col] = (row == col) ? 0.0f : acc[mt][nt][t];
      }
}

// ---------------------------------------------------------------- up path / output
__global__ void k_scatter_add(float* __restrict__ tmp, const float* __restrict__ xl,
                              const int* __restrict__ perm, int k) {
  int i = blockIdx.x * blockDim.x + threadIdx.x;
  if (i >= k * HIDV) return;
  int r = i >> 5, f = i & 31;
  tmp[(size_t)perm[r] * HIDV + f] += xl[i];   // perm entries unique: no atomics needed
}

__global__ void k_log_softmax16(float* __restrict__ io, int n) {
  int i = blockIdx.x * blockDim.x + threadIdx.x;
  if (i >= n) return;
  float* row = io + (size_t)i * NCLSV;
  float m = row[0];
  for (int f = 1; f < NCLSV; ++f) m = fmaxf(m, row[f]);
  float s = 0.0f;
  for (int f = 0; f < NCLSV; ++f) s += expf(row[f] - m);
  float lse = m + logf(s);
  for (int f = 0; f < NCLSV; ++f) row[f] -= lse;
}

// ---------------------------------------------------------------- host helpers
static inline int cdiv(long long a, long long b) { return (int)((a + b - 1) / b); }

struct GcnScratch { float *XW, *XWs, *Y, *dvec; };

static void run_gcn(hipStream_t s, const float* A, int n, const float* Xin, int fin,
                    const float* W, const float* b, int fout, float* Xout, bool relu,
                    GcnScratch sc) {
  k_mm_xw<<<cdiv((long long)n * 32, 256), 256, 0, s>>>(Xin, W, sc.XW, n, fin, fout);
  k_deg_rsqrt<<<cdiv((long long)n * 32, 256), 256, 0, s>>>(A, n, sc.dvec);
  k_scale_rows<<<cdiv((long long)n * fout, 256), 256, 0, s>>>(sc.XW, sc.dvec, sc.XWs, n, fout);
  if ((n & 15) == 0) {
    int waves = n >> 4;
    if (fout == 32)
      k_spmm_tile16<true><<<cdiv((long long)waves * 32, 256), 256, 0, s>>>(A, sc.XWs, sc.Y, n);
    else
      k_spmm_tile16<false><<<cdiv((long long)waves * 32, 256), 256, 0, s>>>(A, sc.XWs, sc.Y, n);
  } else {
    k_spmm_valu<<<cdiv((long long)n * 32, 256), 256, 0, s>>>(A, sc.XWs, sc.Y, n, fout);
  }
  k_combine_gcn<<<cdiv((long long)n * fout, 256), 256, 0, s>>>(sc.Y, sc.XW, sc.dvec, b,
                                                               Xout, n, fout, relu ? 1 : 0);
}

static void run_pool_level(hipStream_t s, const float* Ain, int n, const float* xin,
                           const float* p, int k, int* perm, float* score,
                           _Float16* Ah, _Float16* AhT, float* Apool, float* xpool) {
  k_score<<<cdiv((long long)n * 32, 256), 256, 0, s>>>(xin, p, score, n);
  int P = 1; while (P < n) P <<= 1;
  k_topk_sort<<<1, 1024, 0, s>>>(score, n, P, k, perm);
  k_gather_x<<<cdiv((long long)k * HIDV, 256), 256, 0, s>>>(xin, perm, score, k, xpool);
  int Mp2 = (k + 31) & ~31;                 // A operand rows, padded to 32
  int Np4 = (k + 63) & ~63;                 // B operand rows, padded to 64
  int Kp  = (n + 31) & ~31;                 // reduction dim, padded to 32
  k_gather_rows_f16<<<cdiv((long long)Mp2 * Kp, 256), 256, 0, s>>>(Ain, perm, n, k, Mp2, Kp, Ah);
  k_gather_cols_f16<<<cdiv((long long)Np4 * Kp, 256), 256, 0, s>>>(Ain, perm, n, k, Np4, Kp, AhT);
  int mStrips = Mp2 >> 5, nStrips = Np4 >> 6;
  k_augpool_gemm_f16<<<cdiv((long long)mStrips * nStrips * 32, 256), 256, 0, s>>>(
      Ah, AhT, Apool, k, Kp, mStrips, nStrips);
}

static void run_up(hipStream_t s, const float* res, int n, const float* xlow, int k,
                   const int* perm, float* tmp) {
  k_copy<<<cdiv((long long)n * HIDV, 256), 256, 0, s>>>(res, tmp, n * HIDV);
  k_scatter_add<<<cdiv((long long)k * HIDV, 256), 256, 0, s>>>(tmp, xlow, perm, k);
}

// ---------------------------------------------------------------- entry point
extern "C" void kernel_launch(void* const* d_in, const int* in_sizes, int n_in,
                              void* d_out, int out_size, void* d_ws, size_t ws_size,
                              hipStream_t stream) {
  (void)in_sizes; (void)n_in; (void)out_size; (void)ws_size;

  const float*     x  = (const float*)d_in[0];
  const long long* ei = (const long long*)d_in[1];   // int64 edge_index (2 x E)
  const float *W0 = (const float*)d_in[2],  *b0 = (const float*)d_in[3];
  const float *W1 = (const float*)d_in[4],  *b1 = (const float*)d_in[5];
  const float *W2 = (const float*)d_in[6],  *b2 = (const float*)d_in[7];
  const float *W3 = (const float*)d_in[8],  *b3 = (const float*)d_in[9];
  const float *U0 = (const float*)d_in[10], *c0 = (const float*)d_in[11];
  const float *U1 = (const float*)d_in[12], *c1 = (const float*)d_in[13];
  const float *U2 = (const float*)d_in[14], *c2 = (const float*)d_in[15];
  const float *p0 = (const float*)d_in[16], *p1 = (const float*)d_in[17];
  const float *p2 = (const float*)d_in[18];

  // -------- workspace layout (~125 MB; hot set fits in the 192 MB global L2)
  char* ws = (char*)d_ws;
  size_t off = 0;
  auto alloc = [&](size_t bytes) -> void* {
    void* p = ws + off;
    off = (off + bytes + 255) & ~(size_t)255;
    return p;
  };
  float*    A0    = (float*)alloc((size_t)N0V * N0V * 4);          // 67 MB
  _Float16* Ah    = (_Float16*)alloc((size_t)2048 * N0V * 2);      // 16.8 MB (max Mp2*Kp)
  _Float16* AhT   = (_Float16*)alloc((size_t)2048 * N0V * 2);      // 16.8 MB (max Np4*Kp)
  float*    A1    = (float*)alloc((size_t)KL1 * KL1 * 4);          // 16 MB
  float*    A2    = (float*)alloc((size_t)KL2 * KL2 * 4);          // 4 MB
  float*    A3    = (float*)alloc((size_t)KL3 * KL3 * 4);          // 1 MB
  float*    XW    = (float*)alloc((size_t)N0V * HIDV * 4);
  float*    XWs   = (float*)alloc((size_t)N0V * HIDV * 4);
  float*    Y     = (float*)alloc((size_t)N0V * HIDV * 4);
  float*    dvec  = (float*)alloc((size_t)N0V * 4);
  float*    score = (float*)alloc((size_t)N0V * 4);
  int*      perm1 = (int*)alloc((size_t)KL1 * 4);
  int*      perm2 = (int*)alloc((size_t)KL2 * 4);
  int*      perm3 = (int*)alloc((size_t)KL3 * 4);
  float*    x0    = (float*)alloc((size_t)N0V * HIDV * 4);
  float*    xl1   = (float*)alloc((size_t)KL1 * HIDV * 4);
  float*    xl2   = (float*)alloc((size_t)KL2 * HIDV * 4);
  float*    xl3   = (float*)alloc((size_t)KL3 * HIDV * 4);
  float*    xpool = (float*)alloc((size_t)KL1 * HIDV * 4);
  float*    ux2   = (float*)alloc((size_t)KL2 * HIDV * 4);
  float*    ux1   = (float*)alloc((size_t)KL1 * HIDV * 4);
  float*    uptmp = (float*)alloc((size_t)N0V * HIDV * 4);

  GcnScratch sc{XW, XWs, Y, dvec};

  // dense adjacency (self loops dropped)
  k_fill_zero<<<cdiv((long long)N0V * N0V, 256), 256, 0, stream>>>(A0, N0V * N0V);
  k_scatter_edges<<<cdiv(NEV, 256), 256, 0, stream>>>(A0, ei, NEV, N0V);

  // down path
  run_gcn(stream, A0, N0V, x, FINV, W0, b0, HIDV, x0, true, sc);

  run_pool_level(stream, A0, N0V, x0, p0, KL1, perm1, score, Ah, AhT, A1, xpool);
  run_gcn(stream, A1, KL1, xpool, HIDV, W1, b1, HIDV, xl1, true, sc);

  run_pool_level(stream, A1, KL1, xl1, p1, KL2, perm2, score, Ah, AhT, A2, xpool);
  run_gcn(stream, A2, KL2, xpool, HIDV, W2, b2, HIDV, xl2, true, sc);

  run_pool_level(stream, A2, KL2, xl2, p2, KL3, perm3, score, Ah, AhT, A3, xpool);
  run_gcn(stream, A3, KL3, xpool, HIDV, W3, b3, HIDV, xl3, true, sc);

  // up path
  run_up(stream, xl2, KL2, xl3, KL3, perm3, uptmp);
  run_gcn(stream, A2, KL2, uptmp, HIDV, U0, c0, HIDV, ux2, true, sc);

  run_up(stream, xl1, KL1, ux2, KL2, perm2, uptmp);
  run_gcn(stream, A1, KL1, uptmp, HIDV, U1, c1, HIDV, ux1, true, sc);

  run_up(stream, x0, N0V, ux1, KL1, perm1, uptmp);
  run_gcn(stream, A0, N0V, uptmp, HIDV, U2, c2, NCLSV, (float*)d_out, false, sc);

  k_log_softmax16<<<cdiv(N0V, 256), 256, 0, stream>>>((float*)d_out, N0V);
}
